// InterestEvolving_78752520339735
// MI455X (gfx1250) — matrix operand order repaired and checked by reading, not compiled, not run on hardware
//
#include <hip/hip_runtime.h>
#include <hip/hip_bf16.h>

// ---------------------------------------------------------------------------
// DIN attention (WMMA f16) + fused AUGRU scan (WMMA f16).
//  - All LDS tiles row-padded to kill 16-way bank conflicts on fragment loads.
//  - Scan keeps recurrent h state in f32 registers (lane-exclusive ownership),
//    with an f16 shadow in LDS for cross-wave A-fragment builds.
//  - keys_t staged as f16 once per step (no per-wave duplicate conversions).
//  - global_prefetch for t+1 keys overlaps HBM latency with step-t WMMAs.
// ---------------------------------------------------------------------------

typedef __attribute__((ext_vector_type(16))) _Float16 v16h;
typedef __attribute__((ext_vector_type(8)))  _Float16 v8h;
typedef __attribute__((ext_vector_type(8)))  float    v8f;

#define B_  2048
#define T_  200
#define H_  128
#define A0_ 64
#define A1_ 16

// padded LDS row strides (in halves): row base banks distinct for rows 0..15
#define W0S 520   // 1040 B rows (64x512 tile)
#define W1S 72    // 144 B rows  (16x64 tile)
#define A1S 72    // 144 B rows  (16x64 tile)
#define WS_ 136   // 272 B rows  (384x128 / 16x128 tiles)

__device__ __forceinline__ float sigf(float x) { return 1.0f / (1.0f + __expf(-x)); }
__device__ __forceinline__ float tanhfast(float x) { return 2.0f * sigf(2.0f * x) - 1.0f; }

// load 8 consecutive floats (global)
__device__ __forceinline__ void load8(const float* p, float* o) {
  float4 a = *(const float4*)(p);
  float4 b = *(const float4*)(p + 4);
  o[0]=a.x; o[1]=a.y; o[2]=a.z; o[3]=a.w;
  o[4]=b.x; o[5]=b.y; o[6]=b.z; o[7]=b.w;
}

// Build a 16x32 f16 A-fragment slice from an f16 LDS row.
// Contiguous k order (this code's choice, not the ISA table order): lane(0-15)=row, elements e<8 -> K = off+e ; e>=8 -> K = off+8+(e-8)
// where off = 32*ks + 16*sel is folded into `off` by the caller.
__device__ __forceinline__ v16h frag_f16(const _Float16* row, int off) {
  v8h lo = *(const v8h*)(row + off);
  v8h hi = *(const v8h*)(row + off + 8);
  v16h f;
#pragma unroll
  for (int e = 0; e < 8; ++e) { f[e] = lo[e]; f[e+8] = hi[e]; }
  return f;
}

// ---------------------------------------------------------------------------
// Kernel 1: attention scores att[B,T] = masked( MLP([q,k,q-k,q*k]) )
// 8 waves/block, each wave owns a 16-row M-tile of the [B*T, 512] GEMM.
// ---------------------------------------------------------------------------
__global__ __launch_bounds__(256) void din_att_kernel(
    const float* __restrict__ q,  const float* __restrict__ keys,
    const float* __restrict__ W0, const float* __restrict__ b0,
    const float* __restrict__ W1, const float* __restrict__ b1,
    const float* __restrict__ Wd, const float* __restrict__ bd,
    const int* __restrict__ keys_length, float* __restrict__ att)
{
  extern __shared__ char smem[];
  _Float16* W0h = (_Float16*)smem;                  // 64 x W0S halves (66560 B)
  _Float16* W1h = (_Float16*)(smem + 66560);        // 16 x W1S halves (2304 B)
  _Float16* a1b = (_Float16*)(smem + 68864);        // 8 waves x 16 x A1S halves

  const int tid = threadIdx.x;
  for (int i = tid; i < A0_ * 4 * H_; i += 256) {
    int o = i >> 9, k = i & 511;
    W0h[o * W0S + k] = (_Float16)W0[i];
  }
  for (int i = tid; i < A1_ * A0_; i += 256) {
    int o = i >> 6, k = i & 63;
    W1h[o * W1S + k] = (_Float16)W1[i];
  }
  __syncthreads();

  const int wave = tid >> 5, lane = tid & 31;
  const int nl = lane & 15, sel = lane >> 4;
  const int rowbase = (blockIdx.x * 8 + wave) * 16;
  const int row = rowbase + nl;          // this lane's A-matrix row (b*T + t)
  const int bb = row / T_;
  const float* qrow = q + (size_t)bb * H_;
  const float* krow = keys + (size_t)row * H_;

  float b0v[4];
#pragma unroll
  for (int nt = 0; nt < 4; ++nt) b0v[nt] = b0[nt * 16 + nl];
  const float b1v = b1[nl];
  const float wdv = Wd[nl];
  const float bdv = bd[0];

  v8f acc[4];
#pragma unroll
  for (int nt = 0; nt < 4; ++nt) acc[nt] = {};

  const int jo = 16 * sel;
#pragma unroll
  for (int kt = 0; kt < 16; ++kt) {
    const int r  = kt >> 2;          // concat region: 0=q 1=k 2=q-k 3=q*k
    const int jb = (kt & 3) * 32;    // j offset within H
    float qlo[8], qhi[8], klo[8], khi[8];
    if (r != 1) { load8(qrow + jb + jo, qlo); load8(qrow + jb + 8 + jo, qhi); }
    if (r != 0) { load8(krow + jb + jo, klo); load8(krow + jb + 8 + jo, khi); }
    v16h A;
#pragma unroll
    for (int e = 0; e < 8; ++e) {
      float lo, hi;
      if      (r == 0) { lo = qlo[e];           hi = qhi[e]; }
      else if (r == 1) { lo = klo[e];           hi = khi[e]; }
      else if (r == 2) { lo = qlo[e] - klo[e];  hi = qhi[e] - khi[e]; }
      else             { lo = qlo[e] * klo[e];  hi = qhi[e] * khi[e]; }
      A[e] = (_Float16)lo; A[e + 8] = (_Float16)hi;
    }
#pragma unroll
    for (int nt = 0; nt < 4; ++nt) {
      // B 32x16 f16: lane=col o, K = 16*sel + e  -> 16 contiguous halves
      v16h Bf = *(const v16h*)(W0h + (nt * 16 + nl) * W0S + kt * 32 + 16 * sel);
      acc[nt] = __builtin_amdgcn_wmma_f32_16x16x32_f16(
          false, A, false, Bf, (short)0, acc[nt], false, false);
    }
  }

  // layer-1 sigmoid, re-layout C->A through per-wave LDS tile [16][A1S] f16
  _Float16* a1w = a1b + wave * 16 * A1S;
#pragma unroll
  for (int nt = 0; nt < 4; ++nt)
#pragma unroll
    for (int v = 0; v < 8; ++v) {
      int rv = sel * 8 + v;                            // C layout: VGPR v -> row
      a1w[rv * A1S + nt * 16 + nl] = (_Float16)sigf(acc[nt][v] + b0v[nt]);
    }
  // same-wave LDS ops are in order; no barrier needed (wave-private tile)

  v8f acc2 = {};
#pragma unroll
  for (int kt = 0; kt < 2; ++kt) {
    v16h A  = frag_f16(a1w + nl * A1S, kt * 32 + 16 * sel);
    v16h Bf = *(const v16h*)(W1h + nl * W1S + kt * 32 + 16 * sel);
    acc2 = __builtin_amdgcn_wmma_f32_16x16x32_f16(
        false, A, false, Bf, (short)0, acc2, false, false);
  }

  // score[m] = sum_n sig(a2[m][n] + b1[n]) * Wd[n] + bd ; reduce across 16 lanes
#pragma unroll
  for (int v = 0; v < 8; ++v) {
    float p = sigf(acc2[v] + b1v) * wdv;
#pragma unroll
    for (int mm = 1; mm < 16; mm <<= 1) p += __shfl_xor(p, mm, 32);
    if (nl == 0) {
      int grow = rowbase + sel * 8 + v;
      int gb = grow / T_;
      int gt = grow - gb * T_;
      att[grow] = (gt < keys_length[gb]) ? (p + bdv) : 0.0f;
    }
  }
}

// ---------------------------------------------------------------------------
// Kernel 2: fused gi + AUGRU scan. Block owns 16 rows of B; Wih/Whh staged
// once as padded f16 in LDS; h state lives in f32 registers (lane-exclusive),
// with an f16 LDS shadow for cross-wave fragments. 192 WMMA per step.
// att==0 at invalid steps => z==0 => h frozen automatically.
// ---------------------------------------------------------------------------
__global__ __launch_bounds__(256) void augru_kernel(
    const float* __restrict__ keys, const float* __restrict__ Wih,
    const float* __restrict__ Whh,  const float* __restrict__ bih,
    const float* __restrict__ bhh,  const int* __restrict__ keys_length,
    const float* __restrict__ att,  float* __restrict__ out)
{
  extern __shared__ char smem[];
  _Float16* WihH  = (_Float16*)smem;                // 384 x WS_ halves (104448 B)
  _Float16* WhhH  = (_Float16*)(smem + 104448);     // 384 x WS_ halves (104448 B)
  _Float16* kbufH = (_Float16*)(smem + 208896);     // 16 x WS_ halves (4352 B)
  _Float16* hbufH = (_Float16*)(smem + 213248);     // 16 x WS_ halves (4352 B)
  float*    attv  = (float*)(smem + 217600);        // 16 f32

  const int tid = threadIdx.x;
  const int bbase = blockIdx.x * 16;

  for (int i = tid; i < 3 * H_ * H_; i += 256) {
    int o = i >> 7, k = i & 127;
    WihH[o * WS_ + k] = (_Float16)Wih[i];
    WhhH[o * WS_ + k] = (_Float16)Whh[i];
  }
  for (int i = tid; i < 16 * WS_; i += 256) hbufH[i] = (_Float16)0.0f;

  const int wave = tid >> 5, lane = tid & 31;
  const int nl = lane & 15, sel = lane >> 4;
  const int g0 = wave, g1 = wave + 8, g2 = wave + 16;   // matched r/z/n N-tiles
  const float biR = bih[g0 * 16 + nl], biZ = bih[g1 * 16 + nl], biN = bih[g2 * 16 + nl];
  const float bhR = bhh[g0 * 16 + nl], bhZ = bhh[g1 * 16 + nl], bhN = bhh[g2 * 16 + nl];
  const int col = g0 * 16 + nl;          // this lane's exclusive h column

  float hreg[8];                         // master f32 recurrent state (rows sel*8+v)
#pragma unroll
  for (int v = 0; v < 8; ++v) hreg[v] = 0.0f;

  __syncthreads();

  for (int t = 0; t < T_; ++t) {
    // stage keys[:,t,:] tile as f16 + attention values; prefetch t+1 keys
    for (int i = tid; i < 16 * H_; i += 256) {
      int r = i >> 7, j = i & 127;
      const float* src = keys + ((size_t)(bbase + r) * T_ + t) * H_ + j;
      kbufH[r * WS_ + j] = (_Float16)(*src);
      if ((j & 31) == 0 && t + 1 < T_) __builtin_prefetch(src + H_, 0, 0);
    }
    if (tid < 16) attv[tid] = att[(size_t)(bbase + tid) * T_ + t];
    __syncthreads();

    v8f aR = {}, aZ = {}, aN = {}, hR = {}, hZ = {}, hN = {};
#pragma unroll
    for (int ks = 0; ks < 4; ++ks) {          // gi = keys_t @ Wih^T
      v16h A  = frag_f16(kbufH + nl * WS_, ks * 32 + 16 * sel);
      v16h Br = *(const v16h*)(WihH + (g0 * 16 + nl) * WS_ + ks * 32 + 16 * sel);
      v16h Bz = *(const v16h*)(WihH + (g1 * 16 + nl) * WS_ + ks * 32 + 16 * sel);
      v16h Bn = *(const v16h*)(WihH + (g2 * 16 + nl) * WS_ + ks * 32 + 16 * sel);
      aR = __builtin_amdgcn_wmma_f32_16x16x32_f16(false, A, false, Br, (short)0, aR, false, false);
      aZ = __builtin_amdgcn_wmma_f32_16x16x32_f16(false, A, false, Bz, (short)0, aZ, false, false);
      aN = __builtin_amdgcn_wmma_f32_16x16x32_f16(false, A, false, Bn, (short)0, aN, false, false);
    }
#pragma unroll
    for (int ks = 0; ks < 4; ++ks) {          // gh = h @ Whh^T
      v16h A  = frag_f16(hbufH + nl * WS_, ks * 32 + 16 * sel);
      v16h Br = *(const v16h*)(WhhH + (g0 * 16 + nl) * WS_ + ks * 32 + 16 * sel);
      v16h Bz = *(const v16h*)(WhhH + (g1 * 16 + nl) * WS_ + ks * 32 + 16 * sel);
      v16h Bn = *(const v16h*)(WhhH + (g2 * 16 + nl) * WS_ + ks * 32 + 16 * sel);
      hR = __builtin_amdgcn_wmma_f32_16x16x32_f16(false, A, false, Br, (short)0, hR, false, false);
      hZ = __builtin_amdgcn_wmma_f32_16x16x32_f16(false, A, false, Bz, (short)0, hZ, false, false);
      hN = __builtin_amdgcn_wmma_f32_16x16x32_f16(false, A, false, Bn, (short)0, hN, false, false);
    }

#pragma unroll
    for (int v = 0; v < 8; ++v) {
      int rv = sel * 8 + v;                   // C layout row for this half-wave
      float r = sigf(aR[v] + biR + hR[v] + bhR);
      float z = sigf(aZ[v] + biZ + hZ[v] + bhZ);
      float n = tanhfast(aN[v] + biN + r * (hN[v] + bhN));
      z *= attv[rv];                          // attention-modulated gate (0 if invalid)
      hreg[v] = hreg[v] + z * (n - hreg[v]);  // f32 register recurrence
    }
    __syncthreads();                          // all frag reads of hbufH done
#pragma unroll
    for (int v = 0; v < 8; ++v)
      hbufH[(sel * 8 + v) * WS_ + col] = (_Float16)hreg[v];
    // next iteration's post-stage barrier separates these writes from reads
  }

  // output straight from registers
  const float keep = (keys_length[bbase + sel * 8] > 0) ? 1.0f : 0.0f; // dummy to keep shape
#pragma unroll
  for (int v = 0; v < 8; ++v) {
    int gb = bbase + sel * 8 + v;
    out[(size_t)gb * H_ + col] = hreg[v] * ((keys_length[gb] > 0) ? 1.0f : 0.0f);
  }
  (void)keep;
}

// ---------------------------------------------------------------------------
extern "C" void kernel_launch(void* const* d_in, const int* in_sizes, int n_in,
                              void* d_out, int out_size, void* d_ws, size_t ws_size,
                              hipStream_t stream) {
  (void)in_sizes; (void)n_in; (void)out_size; (void)ws_size;
  const float* q    = (const float*)d_in[0];
  const float* keys = (const float*)d_in[1];
  const float* W0   = (const float*)d_in[2];
  const float* b0   = (const float*)d_in[3];
  const float* W1   = (const float*)d_in[4];
  const float* b1   = (const float*)d_in[5];
  const float* Wd   = (const float*)d_in[6];
  const float* bd   = (const float*)d_in[7];
  const float* Wih  = (const float*)d_in[8];
  const float* Whh  = (const float*)d_in[9];
  const float* bih  = (const float*)d_in[10];
  const float* bhh  = (const float*)d_in[11];
  const int*   klen = (const int*)d_in[12];
  float* out = (float*)d_out;
  float* att = (float*)d_ws;                  // B*T floats of scratch

  const int SMEM1 = 66560 + 2304 + 8 * 16 * A1S * 2;      // 87296 B
  const int SMEM2 = 104448 * 2 + 4352 * 2 + 64;           // 217664 B

  (void)hipFuncSetAttribute(reinterpret_cast<const void*>(din_att_kernel),
                            hipFuncAttributeMaxDynamicSharedMemorySize, SMEM1);
  (void)hipFuncSetAttribute(reinterpret_cast<const void*>(augru_kernel),
                            hipFuncAttributeMaxDynamicSharedMemorySize, SMEM2);

  // 409600 rows / (8 waves * 16 rows) = 3200 blocks
  din_att_kernel<<<(B_ * T_) / 128, 256, SMEM1, stream>>>(
      q, keys, W0, b0, W1, b1, Wd, bd, klen, att);
  // 2048 rows / 16 = 128 blocks
  augru_kernel<<<B_ / 16, 256, SMEM2, stream>>>(
      keys, Wih, Whh, bih, bhh, klen, att, out);
}
